// Attention_2087354105914
// MI455X (gfx1250) — compile-verified, matrix operand order checked by reading, not run on hardware
//
#include <hip/hip_runtime.h>
#include <hip/hip_bf16.h>

// ---------------- WMMA helpers (gfx1250, wave32, 16x16x32 f16 -> f32) --------
typedef __attribute__((ext_vector_type(16))) _Float16 v16h;
typedef __attribute__((ext_vector_type(8)))  _Float16 v8h;
typedef __attribute__((ext_vector_type(8)))  float    v8f;
typedef __attribute__((ext_vector_type(4)))  float    v4f;

__device__ __forceinline__ v8f wmma_f16(v16h a, v16h b, v8f c) {
  // (neg_a, A, neg_b, B, c_mod, C, reuse_a, reuse_b)
  return __builtin_amdgcn_wmma_f32_16x16x32_f16(false, a, false, b, (short)0, c,
                                                false, false);
}

// Fragment loader for BOTH operands:
//  - A operand from row-major [M][K] storage (ld = K stride)
//  - B operand from [N][K] storage (each math-column contiguous)
// ISA layout: lanes 0-15 -> row/col 0-15, K {0..7,16..23};
//             lanes 16-31 -> row/col 0-15, K {8..15,24..31}.
// Each lane: two contiguous 16B vector loads -> global_load_b128/ds_load_b128.
__device__ __forceinline__ v16h load_frag(const _Float16* base, int ld) {
  int lane = threadIdx.x & 31;
  int row  = lane & 15;
  int k0   = (lane & 16) ? 8 : 0;
  const _Float16* p = base + (size_t)row * ld + k0;
  v8h lo = *(const v8h*)(p);
  v8h hi = *(const v8h*)(p + 16);
  return __builtin_shufflevector(lo, hi, 0, 1, 2, 3, 4, 5, 6, 7,
                                 8, 9, 10, 11, 12, 13, 14, 15);
}

// ---------------- problem constants ------------------------------------------
#define BB    32
#define CC    128
#define HH    56
#define NN    (HH * HH)     // 3136
#define NK    225           // 15*15 reduced spatial
#define NKP   256           // padded to multiple of 32 for WMMA K loops
#define HEADS 2
#define DH    64            // head dim

// LDS row pads (halves). 72*2=144B and 264*2=528B keep 16B alignment and give
// conflict-free bank patterns for 16-lane b128 fragment loads.
#define KROW  72
#define VROW  264
#define PROW  264

// ---------------- x transpose: [B][C][N] f32 -> xt [B][N][C] f16 --------------
__global__ __launch_bounds__(256) void xpose_kernel(const float* __restrict__ x,
                                                    _Float16* __restrict__ xt) {
  __shared__ _Float16 tile[32][33];
  int tn = blockIdx.x % (NN / 32);                  // 98 tiles along N
  int tc = (blockIdx.x / (NN / 32)) % (CC / 32);    // 4 tiles along C
  int b  = blockIdx.x / ((NN / 32) * (CC / 32));
  int tx = threadIdx.x & 31;
  int ty = threadIdx.x >> 5;                        // 0..7
  const float* xb = x + (size_t)b * CC * NN;
#pragma unroll
  for (int j = 0; j < 4; ++j) {
    int c = tc * 32 + ty + j * 8;
    int n = tn * 32 + tx;
    tile[ty + j * 8][tx] = (_Float16)xb[(size_t)c * NN + n];
  }
  __syncthreads();
  _Float16* xtb = xt + (size_t)b * NN * CC;
#pragma unroll
  for (int j = 0; j < 4; ++j) {
    int n = tn * 32 + ty + j * 8;
    int c = tc * 32 + tx;
    xtb[(size_t)n * CC + c] = tile[tx][ty + j * 8];
  }
}

// ---------------- weight converters (row-major f16, used as [N][K] B ops) ----
__global__ __launch_bounds__(256) void cvt_w_kernel(const float* __restrict__ qw,
                                                    const float* __restrict__ kvw,
                                                    _Float16* __restrict__ qwh,
                                                    _Float16* __restrict__ kvwh) {
  int i = blockIdx.x * blockDim.x + threadIdx.x;
  if (i < CC * CC) qwh[i] = (_Float16)qw[i];
  if (i < 2 * CC * CC) kvwh[i] = (_Float16)kvw[i];
}

// ---------------- spatial reduction path (tiny; scalar) -----------------------
__global__ __launch_bounds__(256) void sr_stage1_kernel(
    const float* __restrict__ x, const float* __restrict__ sr1_w,
    const float* __restrict__ g1, const float* __restrict__ b1,
    const float* __restrict__ m1, const float* __restrict__ v1,
    const float* __restrict__ sr2_w, const float* __restrict__ g2,
    const float* __restrict__ b2, const float* __restrict__ m2,
    const float* __restrict__ v2, float* __restrict__ kv1) {
  int idx = blockIdx.x * blockDim.x + threadIdx.x;
  if (idx >= BB * CC * NK) return;
  int m = idx % NK;
  int c = (idx / NK) % CC;
  int b = idx / (NK * CC);
  int oy = m / 15, ox = m % 15;
  const float* xp = x + ((size_t)b * CC + c) * NN;
  const float* wp = sr1_w + c * 16;
  float acc = 0.f;
#pragma unroll
  for (int ky = 0; ky < 4; ++ky) {
    int iy = oy * 4 - 2 + ky;
    if (iy < 0 || iy >= HH) continue;
#pragma unroll
    for (int kx = 0; kx < 4; ++kx) {
      int ix = ox * 4 - 2 + kx;
      if (ix < 0 || ix >= HH) continue;
      acc += xp[iy * HH + ix] * wp[ky * 4 + kx];
    }
  }
  float t = (acc - m1[c]) * (g1[c] * rsqrtf(v1[c] + 1e-5f)) + b1[c];
  t = fmaxf(t, 0.f);
  t = t * sr2_w[c];
  t = (t - m2[c]) * (g2[c] * rsqrtf(v2[c] + 1e-5f)) + b2[c];
  kv1[idx] = t;
}

// stage2: depthwise 3x3 pad1 + bias + residual -> kvt f16 [B][NKP][C], zero pad
__global__ __launch_bounds__(256) void sr_stage2_kernel(
    const float* __restrict__ kv1, const float* __restrict__ local_w,
    const float* __restrict__ local_b, _Float16* __restrict__ kvt) {
  int idx = blockIdx.x * blockDim.x + threadIdx.x;
  if (idx >= BB * NKP * CC) return;
  int c  = idx % CC;
  int mp = (idx / CC) % NKP;
  int b  = idx / (CC * NKP);
  float val = 0.f;
  if (mp < NK) {
    int oy = mp / 15, ox = mp % 15;
    const float* kp = kv1 + ((size_t)b * CC + c) * NK;
    const float* wp = local_w + c * 9;
    float acc = 0.f;
#pragma unroll
    for (int ky = 0; ky < 3; ++ky) {
      int iy = oy - 1 + ky;
      if (iy < 0 || iy >= 15) continue;
#pragma unroll
      for (int kx = 0; kx < 3; ++kx) {
        int ix = ox - 1 + kx;
        if (ix < 0 || ix >= 15) continue;
        acc += kp[iy * 15 + ix] * wp[ky * 3 + kx];
      }
    }
    val = acc + local_b[c] + kp[oy * 15 + ox];
  }
  kvt[idx] = (_Float16)val;    // [b][mp][c]
}

// ---------------- kv projection: D[m][c_out] = kvt[m][c] x kv_w[c_out][c]^T ---
__global__ __launch_bounds__(256) void kvproj_gemm_kernel(
    const _Float16* __restrict__ kvwh, const _Float16* __restrict__ kvt,
    const float* __restrict__ kv_b, _Float16* __restrict__ kt,
    _Float16* __restrict__ varr) {
  int wid = blockIdx.x * 8 + (threadIdx.x >> 5);
  int b  = wid >> 8;
  int t  = wid & 255;
  int mt = t >> 4;            // m tile 0..15
  int nt = t & 15;            // c_out tile 0..15
  const _Float16* Abase = kvt + ((size_t)b * NKP + mt * 16) * CC;
  const _Float16* Bbase = kvwh + (size_t)nt * 16 * CC;
  v8f acc = {};
#pragma unroll
  for (int kk = 0; kk < CC / 32; ++kk) {
    acc = wmma_f16(load_frag(Abase + kk * 32, CC),
                   load_frag(Bbase + kk * 32, CC), acc);
  }
  int lane = threadIdx.x & 31;
  int col  = lane & 15;
  int row0 = (lane & 16) ? 8 : 0;
  int c = nt * 16 + col;      // uniform K/V split per tile (no divergence)
  if (c < CC) {
#pragma unroll
    for (int r = 0; r < 8; ++r) {
      int m = mt * 16 + row0 + r;
      kt[((size_t)b * NKP + m) * CC + c] = (_Float16)(acc[r] + kv_b[c]);
    }
  } else {
#pragma unroll
    for (int r = 0; r < 8; ++r) {
      int m = mt * 16 + row0 + r;
      varr[((size_t)b * CC + (c - CC)) * NKP + m] = (_Float16)(acc[r] + kv_b[c]);
    }
  }
}

// ---------------- q projection: D[n][c_out] = xt[n][c] x q_w[c_out][c]^T ------
__global__ __launch_bounds__(256) void qproj_gemm_kernel(
    const _Float16* __restrict__ qwh, const _Float16* __restrict__ xt,
    const float* __restrict__ q_b, _Float16* __restrict__ qt) {
  int wid = blockIdx.x * 8 + (threadIdx.x >> 5);
  int b  = wid / ((NN / 16) * 8);
  int t  = wid % ((NN / 16) * 8);
  int mt = t >> 3;            // n tile 0..195
  int nt = t & 7;             // c_out tile 0..7
  const _Float16* Abase = xt + ((size_t)b * NN + mt * 16) * CC;
  const _Float16* Bbase = qwh + (size_t)nt * 16 * CC;
  v8f acc = {};
#pragma unroll
  for (int kk = 0; kk < CC / 32; ++kk) {
    acc = wmma_f16(load_frag(Abase + kk * 32, CC),
                   load_frag(Bbase + kk * 32, CC), acc);
  }
  int lane = threadIdx.x & 31;
  int col  = lane & 15;
  int row0 = (lane & 16) ? 8 : 0;
  int c = nt * 16 + col;
#pragma unroll
  for (int r = 0; r < 8; ++r) {
    int n = mt * 16 + row0 + r;
    qt[((size_t)b * NN + n) * CC + c] = (_Float16)(acc[r] + q_b[c]);
  }
}

// ---------------- fused attention with KV resident in LDS ---------------------
// 256 blocks = (b, head, quarter of N). 4 waves/block stage K (256x64) and
// V (64x256) into LDS once, then each wave loops over ~12 query tiles.
#define QTILES 49   // 196/4 query tiles per block
__global__ __launch_bounds__(128) void attn_kernel(
    const _Float16* __restrict__ qt, const _Float16* __restrict__ kt,
    const _Float16* __restrict__ varr, float* __restrict__ out) {
  __shared__ _Float16 ldsK[NKP * KROW];      // [m][dd], padded rows
  __shared__ _Float16 ldsV[DH * VROW];       // [dd][m], padded rows
  __shared__ _Float16 ldsP[4][16 * PROW];    // per-wave P tile

  int blk = blockIdx.x;                      // BB * HEADS * 4
  int b   = blk / (HEADS * 4);
  int rem = blk % (HEADS * 4);
  int hh  = rem / 4;
  int qtr = rem % 4;

  int tid  = threadIdx.x;
  int w    = tid >> 5;
  int lane = tid & 31;
  int col  = lane & 15;
  int row0 = (lane & 16) ? 8 : 0;

  // ---- stage K: ldsK[m][0..63] = kt[b][m][hh*64 + dd] (2 rows per thread)
  {
    const _Float16* ks = kt + (size_t)b * NKP * CC + hh * DH;
#pragma unroll
    for (int it = 0; it < 2; ++it) {
      int m = tid + it * 128;
      const v8h* s = (const v8h*)(ks + (size_t)m * CC);
      v8h* d = (v8h*)(&ldsK[m * KROW]);
#pragma unroll
      for (int u = 0; u < 8; ++u) d[u] = s[u];
    }
  }
  // ---- stage V: ldsV[dd][0..255] = varr[b][hh*64+dd][m] (half row per thread)
  {
    int row = tid & 63;
    int seg = tid >> 6;                      // 0..1, 128 halves each
    const v8h* s = (const v8h*)(varr + ((size_t)b * CC + hh * DH + row) * NKP +
                                seg * 128);
    v8h* d = (v8h*)(&ldsV[row * VROW + seg * 128]);
#pragma unroll
    for (int u = 0; u < 16; ++u) d[u] = s[u];
  }
  __syncthreads();

  const float scale = 0.125f;  // 64^-0.5
  float* ob = out + (size_t)b * CC * NN;
  const _Float16* qhead = qt + (size_t)b * NN * CC + hh * DH;

  for (int i = w; i < QTILES; i += 4) {
    int nt = qtr * QTILES + i;
    int n0 = nt * 16;

    // prefetch next tile's Q rows while this tile computes
    {
      int ntn = nt + 4 < 196 ? nt + 4 : nt;
      __builtin_prefetch(qhead + (size_t)(ntn * 16 + (lane & 15)) * CC, 0, 3);
    }

    // Q tile A-operands (K = 64 -> two 32-wide steps)
    const _Float16* qbase = qhead + (size_t)n0 * CC;
    v16h a0 = load_frag(qbase, CC);
    v16h a1 = load_frag(qbase + 32, CC);

    // S = Q K^T over 16 m tiles (B fragments from LDS, conflict-free)
    v8f s[NKP / 16];
#pragma unroll
    for (int j = 0; j < NKP / 16; ++j) {
      v8f acc = {};
      acc = wmma_f16(a0, load_frag(&ldsK[j * 16 * KROW], KROW), acc);
      acc = wmma_f16(a1, load_frag(&ldsK[j * 16 * KROW + 32], KROW), acc);
      s[j] = acc;
    }

    // softmax across m (225 valid, rest -> -inf); row lives in 16 lanes of one
    // half-wave (C layout) -> xor-shuffle reduce over 16 lanes.
#pragma unroll
    for (int r = 0; r < 8; ++r) {
      float mx = -3.0e38f;
#pragma unroll
      for (int j = 0; j < NKP / 16; ++j) {
        int m = j * 16 + col;
        float vsc = (m < NK) ? s[j][r] * scale : -3.0e38f;
        s[j][r] = vsc;
        mx = fmaxf(mx, vsc);
      }
#pragma unroll
      for (int ofs = 1; ofs < 16; ofs <<= 1) mx = fmaxf(mx, __shfl_xor(mx, ofs, 32));
      float sum = 0.f;
#pragma unroll
      for (int j = 0; j < NKP / 16; ++j) {
        float e = __expf(s[j][r] - mx);
        s[j][r] = e;
        sum += e;
      }
#pragma unroll
      for (int ofs = 1; ofs < 16; ofs <<= 1) sum += __shfl_xor(sum, ofs, 32);
      float inv = 1.0f / sum;
#pragma unroll
      for (int j = 0; j < NKP / 16; ++j) {
        ldsP[w][(row0 + r) * PROW + j * 16 + col] = (_Float16)(s[j][r] * inv);
      }
    }
    // same-wave LDS RAW (P store -> P load): hardware keeps DS ops in order,
    // compiler inserts s_wait_dscnt; no cross-wave sharing of ldsP[w].

    // O = P V  (K = 256 -> 8 steps; 4 dd column tiles), all operands in LDS
    v8f o0 = {}, o1 = {}, o2 = {}, o3 = {};
#pragma unroll
    for (int kk = 0; kk < NKP / 32; ++kk) {
      v16h pa = load_frag(&ldsP[w][kk * 32], PROW);
      o0 = wmma_f16(pa, load_frag(&ldsV[0 * 16 * VROW + kk * 32], VROW), o0);
      o1 = wmma_f16(pa, load_frag(&ldsV[1 * 16 * VROW + kk * 32], VROW), o1);
      o2 = wmma_f16(pa, load_frag(&ldsV[2 * 16 * VROW + kk * 32], VROW), o2);
      o3 = wmma_f16(pa, load_frag(&ldsV[3 * 16 * VROW + kk * 32], VROW), o3);
    }

    // store: out[b][hh*64 + dt*16 + col][n0 + row0 + 0..7], 2x b128 per column
    v8f oarr[4] = {o0, o1, o2, o3};
#pragma unroll
    for (int dt = 0; dt < 4; ++dt) {
      int ch = hh * DH + dt * 16 + col;
      float* p = &ob[(size_t)ch * NN + n0 + row0];
      *(v4f*)(p)     = __builtin_shufflevector(oarr[dt], oarr[dt], 0, 1, 2, 3);
      *(v4f*)(p + 4) = __builtin_shufflevector(oarr[dt], oarr[dt], 4, 5, 6, 7);
    }
  }
}

// ---------------- launch ------------------------------------------------------
extern "C" void kernel_launch(void* const* d_in, const int* in_sizes, int n_in,
                              void* d_out, int out_size, void* d_ws, size_t ws_size,
                              hipStream_t stream) {
  const float* x      = (const float*)d_in[0];
  const float* q_w    = (const float*)d_in[1];
  const float* q_b    = (const float*)d_in[2];
  const float* kv_w   = (const float*)d_in[3];
  const float* kv_b   = (const float*)d_in[4];
  const float* sr1_w  = (const float*)d_in[5];
  const float* sr1_g  = (const float*)d_in[6];
  const float* sr1_b  = (const float*)d_in[7];
  const float* sr1_m  = (const float*)d_in[8];
  const float* sr1_v  = (const float*)d_in[9];
  const float* sr2_w  = (const float*)d_in[10];
  const float* sr2_g  = (const float*)d_in[11];
  const float* sr2_b  = (const float*)d_in[12];
  const float* sr2_m  = (const float*)d_in[13];
  const float* sr2_v  = (const float*)d_in[14];
  const float* localw = (const float*)d_in[15];
  const float* localb = (const float*)d_in[16];
  (void)in_sizes; (void)n_in; (void)out_size; (void)ws_size;

  char* ws = (char*)d_ws;
  size_t off = 0;
  auto take = [&](size_t bytes) -> char* {
    char* p = ws + off;
    off = (off + bytes + 255) & ~(size_t)255;
    return p;
  };
  _Float16* xt   = (_Float16*)take((size_t)BB * NN * CC * 2);   // x^T f16
  _Float16* qt   = (_Float16*)take((size_t)BB * NN * CC * 2);   // Q [b][n][c]
  float*    kv1  = (float*)   take((size_t)BB * CC * NK * 4);   // SR stage1
  _Float16* kvt  = (_Float16*)take((size_t)BB * NKP * CC * 2);  // KV in [b][m][c]
  _Float16* kt   = (_Float16*)take((size_t)BB * NKP * CC * 2);  // K  [b][m][c]
  _Float16* varr = (_Float16*)take((size_t)BB * CC * NKP * 2);  // V  [b][c][m]
  _Float16* qwh  = (_Float16*)take((size_t)CC * CC * 2);
  _Float16* kvwh = (_Float16*)take((size_t)2 * CC * CC * 2);

  xpose_kernel<<<BB * (NN / 32) * (CC / 32), 256, 0, stream>>>(x, xt);
  cvt_w_kernel<<<(2 * CC * CC + 255) / 256, 256, 0, stream>>>(q_w, kv_w, qwh, kvwh);
  sr_stage1_kernel<<<(BB * CC * NK + 255) / 256, 256, 0, stream>>>(
      x, sr1_w, sr1_g, sr1_b, sr1_m, sr1_v, sr2_w, sr2_g, sr2_b, sr2_m, sr2_v, kv1);
  sr_stage2_kernel<<<(BB * NKP * CC) / 256, 256, 0, stream>>>(
      kv1, localw, localb, kvt);
  kvproj_gemm_kernel<<<(BB * 256) / 8, 256, 0, stream>>>(kvwh, kvt, kv_b, kt, varr);
  qproj_gemm_kernel<<<(BB * (NN / 16) * 8) / 8, 256, 0, stream>>>(qwh, xt, q_b, qt);
  attn_kernel<<<BB * HEADS * 4, 128, 0, stream>>>(qt, kt, varr, (float*)d_out);
}
